// BasicTransformerBlock_60284160967123
// MI455X (gfx1250) — matrix operand order repaired and checked by reading, not compiled, not run on hardware
//
#include <hip/hip_runtime.h>

typedef unsigned short u16;
typedef __attribute__((ext_vector_type(16))) __bf16 v16bf;
typedef __attribute__((ext_vector_type(8)))  __bf16 v8bf;
typedef __attribute__((ext_vector_type(8)))  float  v8f;

// ---------------- numeric helpers ----------------
__device__ __forceinline__ u16 f2bf(float f) {
    unsigned u = __float_as_uint(f);
    unsigned r = u + 0x7FFFu + ((u >> 16) & 1u);   // round-to-nearest-even
    return (u16)(r >> 16);
}
__device__ __forceinline__ float bf2f(u16 v) {
    return __uint_as_float(((unsigned)v) << 16);
}

// A-fragment (16x32 bf16): lane r=row, half h; elements = K[h*16 .. h*16+15] (contiguous)
__device__ __forceinline__ v16bf load_frag_a(const u16* base, int h) {
    v16bf a;
    const v8bf* p0 = (const v8bf*)(base + h * 16);
    const v8bf* p1 = (const v8bf*)(base + h * 16 + 8);
#pragma unroll
    for (int e = 0; e < 8; ++e) { a[e] = (*p0)[e]; a[8 + e] = (*p1)[e]; }
    return a;
}
// B-fragment (32x16 bf16): lane c=col, half h; elements = K[h*16 .. h*16+15] (contiguous)
__device__ __forceinline__ v16bf load_frag_b(const u16* base, int h) {
    v16bf b;
    const v8bf* p0 = (const v8bf*)(base + h * 16);
    const v8bf* p1 = (const v8bf*)(base + h * 16 + 8);
#pragma unroll
    for (int e = 0; e < 8; ++e) { b[e] = (*p0)[e]; b[8 + e] = (*p1)[e]; }
    return b;
}
__device__ __forceinline__ v8f wmma_bf16(v16bf a, v16bf b, v8f c) {
    return __builtin_amdgcn_wmma_f32_16x16x32_bf16(false, a, false, b, (short)0, c,
                                                   false, false);
}
__device__ __forceinline__ float gelu_exact(float x) {
    return 0.5f * x * (1.0f + erff(x * 0.70710678118654752f));
}

// ---------------- elementwise converters ----------------
__global__ void cvt_f32_bf16_kernel(const float* __restrict__ in,
                                    u16* __restrict__ out, int n) {
    int i = blockIdx.x * 256 + threadIdx.x;
    if (i < n) out[i] = f2bf(in[i]);
}

// W [K][N] fp32 (row-major)  ->  Wt [N][K] bf16
__global__ void wt_kernel(const float* __restrict__ W, u16* __restrict__ Wt,
                          int K, int N) {
    int i = blockIdx.x * 256 + threadIdx.x;
    if (i >= K * N) return;
    int k = i / N, n = i - k * N;
    Wt[(size_t)n * K + k] = f2bf(W[i]);
}

// ---------------- generic WMMA bf16 GEMM ----------------
// C[M x N] = A[M x K](bf16,row-major) * Bt[N x K](bf16,row-major, i.e. B^T) (+epilogue)
// grid = (ceil(M/64), N/64), block = 128 (4 waves). Wave w computes a 16x64 strip:
// rows [bx*64 + w*16, +16), cols [by*64, +64) as 4 accumulators; one A fragment is
// reused by 4 independent WMMAs per k-step (no D->A/B hazards, deep pipelining).
// modes:
//  1: QK scatter: out bf16 at ((b*8+hh)*p1 + r)*96 + d  (b=row/p0, r=row%p0, hh=col/80, d=col%80)
//  2: V transpose scatter: out bf16 at ((b*8+hh)*96 + d)*p1 + r
//  3: fp32 out = acc + bias[col] + resid[row*N+col]
//  4: bf16 out[row*N+col] = gelu(acc + bias[col])
//  5: bf16 out[row*N+col] = (acc + bias[col]) * bf2f(out[row*N+col])
__global__ __launch_bounds__(128) void gemm_wmma_kernel(
    const u16* __restrict__ A, const u16* __restrict__ Bt, int M, int K, int mode,
    u16* __restrict__ outb, float* __restrict__ outf, const float* __restrict__ bias,
    const float* __restrict__ resid, int p0, int p1) {
    const int lane = threadIdx.x & 31;
    const int wave = threadIdx.x >> 5;
    const int r = lane & 15;
    const int h = lane >> 4;
    const int rowBase = blockIdx.x * 64 + wave * 16;
    const int colBase = blockIdx.y * 64;
    const int Ncols = gridDim.y * 64;

    const int arow = rowBase + r;
    const bool aval = (arow < M);
    const u16* arp = A + (size_t)(aval ? arow : 0) * K;
    const u16* brp0 = Bt + (size_t)(colBase + r) * K;
    const u16* brp1 = Bt + (size_t)(colBase + 16 + r) * K;
    const u16* brp2 = Bt + (size_t)(colBase + 32 + r) * K;
    const u16* brp3 = Bt + (size_t)(colBase + 48 + r) * K;

    v8f acc[4];
#pragma unroll
    for (int j = 0; j < 4; ++j)
#pragma unroll
        for (int i = 0; i < 8; ++i) acc[j][i] = 0.f;

    if (rowBase + 16 <= M) {
        // fast path: all 16 rows of this wave's strip are in range
        for (int k0 = 0; k0 < K; k0 += 32) {
            v16bf a = load_frag_a(arp + k0, h);
            v16bf b0 = load_frag_b(brp0 + k0, h);
            v16bf b1 = load_frag_b(brp1 + k0, h);
            v16bf b2 = load_frag_b(brp2 + k0, h);
            v16bf b3 = load_frag_b(brp3 + k0, h);
            acc[0] = wmma_bf16(a, b0, acc[0]);
            acc[1] = wmma_bf16(a, b1, acc[1]);
            acc[2] = wmma_bf16(a, b2, acc[2]);
            acc[3] = wmma_bf16(a, b3, acc[3]);
        }
    } else {
        // edge path: per-lane row guard (only last row-tile of M%64 != 0 GEMMs)
        for (int k0 = 0; k0 < K; k0 += 32) {
            v16bf a = load_frag_a(arp + k0, h);
            if (!aval) {
#pragma unroll
                for (int e = 0; e < 16; ++e) a[e] = (__bf16)0.0f;
            }
            v16bf b0 = load_frag_b(brp0 + k0, h);
            v16bf b1 = load_frag_b(brp1 + k0, h);
            v16bf b2 = load_frag_b(brp2 + k0, h);
            v16bf b3 = load_frag_b(brp3 + k0, h);
            acc[0] = wmma_bf16(a, b0, acc[0]);
            acc[1] = wmma_bf16(a, b1, acc[1]);
            acc[2] = wmma_bf16(a, b2, acc[2]);
            acc[3] = wmma_bf16(a, b3, acc[3]);
        }
    }

#pragma unroll
    for (int j = 0; j < 4; ++j) {
        const int col = colBase + j * 16 + r;
#pragma unroll
        for (int i = 0; i < 8; ++i) {
            int row = rowBase + i + 8 * h;
            if (row >= M) continue;
            float v = acc[j][i];
            switch (mode) {
            case 1: {
                int b = row / p0, rr = row - b * p0;
                int hh = col / 80, d = col - hh * 80;
                outb[((size_t)((b * 8 + hh) * p1 + rr)) * 96 + d] = f2bf(v);
                break;
            }
            case 2: {
                int b = row / p0, rr = row - b * p0;
                int hh = col / 80, d = col - hh * 80;
                outb[((size_t)((b * 8 + hh) * 96 + d)) * p1 + rr] = f2bf(v);
                break;
            }
            case 3:
                outf[(size_t)row * Ncols + col] =
                    v + bias[col] + resid[(size_t)row * Ncols + col];
                break;
            case 4:
                outb[(size_t)row * Ncols + col] = f2bf(gelu_exact(v + bias[col]));
                break;
            case 5: {
                size_t idx = (size_t)row * Ncols + col;
                outb[idx] = f2bf((v + bias[col]) * bf2f(outb[idx]));
                break;
            }
            default: break;
            }
        }
    }
}

// ---------------- flash attention (wave32, WMMA) ----------------
// Q: [B*H][1024][96] bf16 ; Kb: [B*H][nkpad][96] bf16 ; Vt: [B*H][96][nkpad] bf16
// out: [B*1024][640] bf16.  grid = (1024/16, B*8), block = 32 (one wave)
__global__ __launch_bounds__(32) void attn_kernel(
    const u16* __restrict__ Q, const u16* __restrict__ Kb, const u16* __restrict__ Vt,
    u16* __restrict__ out, int nkpad, int mkeys, float scale) {
    __shared__ __align__(16) u16 Pl[16 * 32];

    const int lane = threadIdx.x & 31;
    const int r = lane & 15;
    const int h = lane >> 4;
    const int qt = blockIdx.x;
    const int bh = blockIdx.y;
    const int b = bh >> 3;
    const int hh = bh & 7;

    // Q fragments for the 3 K-steps covering dpad=96
    const u16* qrow = Q + ((size_t)bh * 1024 + qt * 16 + r) * 96;
    v16bf aq[3];
#pragma unroll
    for (int s = 0; s < 3; ++s) aq[s] = load_frag_a(qrow + s * 32, h);

    v8f o[6];
    float mrun[8], lrun[8];
#pragma unroll
    for (int t = 0; t < 6; ++t)
#pragma unroll
        for (int i = 0; i < 8; ++i) o[t][i] = 0.f;
#pragma unroll
    for (int i = 0; i < 8; ++i) { mrun[i] = -3.0e38f; lrun[i] = 0.f; }

    const int nch = nkpad >> 5;
    for (int kc = 0; kc < nch; ++kc) {
        // S = Q * K^T for 32 keys (two 16-key subtiles), accumulate over dpad=96
        v8f s0 = {0.f, 0.f, 0.f, 0.f, 0.f, 0.f, 0.f, 0.f};
        v8f s1 = {0.f, 0.f, 0.f, 0.f, 0.f, 0.f, 0.f, 0.f};
        const int key0 = kc * 32 + r;
        const int key1 = kc * 32 + 16 + r;
#pragma unroll
        for (int s = 0; s < 3; ++s) {
            v16bf bk0 = load_frag_b(Kb + ((size_t)bh * nkpad + key0) * 96 + s * 32, h);
            s0 = wmma_bf16(aq[s], bk0, s0);
            v16bf bk1 = load_frag_b(Kb + ((size_t)bh * nkpad + key1) * 96 + s * 32, h);
            s1 = wmma_bf16(aq[s], bk1, s1);
        }

        // online softmax (rows = i + 8h; key dim = across the 16-lane half)
        float corrv[8], p0v[8], p1v[8];
#pragma unroll
        for (int i = 0; i < 8; ++i) {
            float v0 = s0[i] * scale;
            float v1 = s1[i] * scale;
            if (key0 >= mkeys) v0 = -3.0e38f;
            if (key1 >= mkeys) v1 = -3.0e38f;
            float mx = fmaxf(v0, v1);
            mx = fmaxf(mx, __shfl_xor(mx, 1, 16));
            mx = fmaxf(mx, __shfl_xor(mx, 2, 16));
            mx = fmaxf(mx, __shfl_xor(mx, 4, 16));
            mx = fmaxf(mx, __shfl_xor(mx, 8, 16));
            float nm = fmaxf(mrun[i], mx);
            float corr = __expf(mrun[i] - nm);
            float e0 = __expf(v0 - nm);
            float e1 = __expf(v1 - nm);
            float ps = e0 + e1;
            ps += __shfl_xor(ps, 1, 16);
            ps += __shfl_xor(ps, 2, 16);
            ps += __shfl_xor(ps, 4, 16);
            ps += __shfl_xor(ps, 8, 16);
            lrun[i] = lrun[i] * corr + ps;
            mrun[i] = nm;
            corrv[i] = corr;
            p0v[i] = e0;
            p1v[i] = e1;
        }
#pragma unroll
        for (int t = 0; t < 6; ++t)
#pragma unroll
            for (int i = 0; i < 8; ++i) o[t][i] *= corrv[i];

        // Re-layout P (C-layout -> A-layout) through LDS
        __syncthreads();
#pragma unroll
        for (int i = 0; i < 8; ++i) {
            Pl[(i + 8 * h) * 32 + r] = f2bf(p0v[i]);
            Pl[(i + 8 * h) * 32 + 16 + r] = f2bf(p1v[i]);
        }
        __syncthreads();
        v16bf pa = load_frag_a(Pl + r * 32, h);

        // O += P * V  (6 column groups of 16 over dpad=96)
#pragma unroll
        for (int t = 0; t < 6; ++t) {
            v16bf bv =
                load_frag_b(Vt + ((size_t)bh * 96 + t * 16 + r) * nkpad + kc * 32, h);
            o[t] = wmma_bf16(pa, bv, o[t]);
        }
    }

    // normalize + store (strip d-pad)
#pragma unroll
    for (int t = 0; t < 6; ++t) {
        int d = t * 16 + r;
        if (d < 80) {
#pragma unroll
            for (int i = 0; i < 8; ++i) {
                int q = qt * 16 + i + 8 * h;
                float v = o[t][i] / lrun[i];
                out[((size_t)(b * 1024 + q)) * 640 + hh * 80 + d] = f2bf(v);
            }
        }
    }
}

// ---------------- LayerNorm (row of 640) ----------------
__global__ __launch_bounds__(256) void ln_kernel(const float* __restrict__ in,
                                                 const float* __restrict__ g,
                                                 const float* __restrict__ bta,
                                                 float* __restrict__ outf,
                                                 u16* __restrict__ outb) {
    __shared__ float s1[256], s2[256];
    const int row = blockIdx.x;
    const int tid = threadIdx.x;
    const float* x = in + (size_t)row * 640;
    float sum = 0.f, sq = 0.f;
    for (int c = tid; c < 640; c += 256) {
        float v = x[c];
        sum += v;
        sq += v * v;
    }
    s1[tid] = sum;
    s2[tid] = sq;
    __syncthreads();
    for (int st = 128; st > 0; st >>= 1) {
        if (tid < st) {
            s1[tid] += s1[tid + st];
            s2[tid] += s2[tid + st];
        }
        __syncthreads();
    }
    float mu = s1[0] * (1.0f / 640.0f);
    float var = s2[0] * (1.0f / 640.0f) - mu * mu;
    float rstd = rsqrtf(var + 1e-5f);
    for (int c = tid; c < 640; c += 256) {
        float v = (x[c] - mu) * rstd * g[c] + bta[c];
        outf[(size_t)row * 640 + c] = v;
        if (outb) outb[(size_t)row * 640 + c] = f2bf(v);
    }
}

// ---------------- host orchestration ----------------
extern "C" void kernel_launch(void* const* d_in, const int* in_sizes, int n_in,
                              void* d_out, int out_size, void* d_ws, size_t ws_size,
                              hipStream_t stream) {
    (void)in_sizes; (void)n_in; (void)out_size; (void)ws_size;

    constexpr int B = 8, NQ = 1024, D = 640, MC = 77, CD = 768;
    constexpr int T = B * NQ;      // 8192
    constexpr int TC = B * MC;     // 616
    constexpr int HID = 2560;
    const float scale = 0.11180339887498949f;  // 1/sqrt(80)

    const float* x      = (const float*)d_in[0];
    const float* ctx    = (const float*)d_in[1];
    const float* sa_wq  = (const float*)d_in[2];
    const float* sa_wk  = (const float*)d_in[3];
    const float* sa_wv  = (const float*)d_in[4];
    const float* sa_wo  = (const float*)d_in[5];
    const float* sa_bo  = (const float*)d_in[6];
    const float* ca_wq  = (const float*)d_in[7];
    const float* ca_wk  = (const float*)d_in[8];
    const float* ca_wv  = (const float*)d_in[9];
    const float* ca_wo  = (const float*)d_in[10];
    const float* ca_bo  = (const float*)d_in[11];
    const float* ff_w1  = (const float*)d_in[12];
    const float* ff_b1  = (const float*)d_in[13];
    const float* ff_w2  = (const float*)d_in[14];
    const float* ff_b2  = (const float*)d_in[15];
    const float* ln1_g  = (const float*)d_in[16];
    const float* ln1_b  = (const float*)d_in[17];
    const float* ln2_g  = (const float*)d_in[18];
    const float* ln2_b  = (const float*)d_in[19];
    const float* ln3_g  = (const float*)d_in[20];
    const float* ln3_b  = (const float*)d_in[21];

    char* ws = (char*)d_ws;
    size_t off = 0;
    auto alloc = [&](size_t bytes) -> char* {
        char* p = ws + off;
        off += (bytes + 255) & ~(size_t)255;
        return p;
    };
    u16* xb     = (u16*)alloc((size_t)T * D * 2);
    u16* ctxb   = (u16*)alloc((size_t)TC * CD * 2);
    u16* wsaq   = (u16*)alloc((size_t)D * D * 2);
    u16* wsak   = (u16*)alloc((size_t)D * D * 2);
    u16* wsav   = (u16*)alloc((size_t)D * D * 2);
    u16* wsao   = (u16*)alloc((size_t)D * D * 2);
    u16* wcaq   = (u16*)alloc((size_t)D * D * 2);
    u16* wcak   = (u16*)alloc((size_t)D * CD * 2);
    u16* wcav   = (u16*)alloc((size_t)D * CD * 2);
    u16* wcao   = (u16*)alloc((size_t)D * D * 2);
    u16* wff1   = (u16*)alloc((size_t)(2 * HID) * D * 2);
    u16* wff2   = (u16*)alloc((size_t)D * HID * 2);
    u16* Qb     = (u16*)alloc((size_t)B * 8 * NQ * 96 * 2);
    u16* KbSA   = (u16*)alloc((size_t)B * 8 * NQ * 96 * 2);
    u16* VtSA   = (u16*)alloc((size_t)B * 8 * 96 * NQ * 2);
    u16* KbCA   = (u16*)alloc((size_t)B * 8 * 96 * 96 * 2);
    u16* VtCA   = (u16*)alloc((size_t)B * 8 * 96 * 96 * 2);
    u16* att    = (u16*)alloc((size_t)T * D * 2);
    float* res  = (float*)alloc((size_t)T * D * 4);
    float* ybuf = (float*)alloc((size_t)T * D * 4);
    u16* ybb    = (u16*)alloc((size_t)T * D * 2);
    u16* gbuf   = (u16*)alloc((size_t)T * HID * 2);

    // zero the d-pad / key-pad regions (whole buffers, cheap)
    hipMemsetAsync(Qb,   0, (size_t)B * 8 * NQ * 96 * 2, stream);
    hipMemsetAsync(KbSA, 0, (size_t)B * 8 * NQ * 96 * 2, stream);
    hipMemsetAsync(VtSA, 0, (size_t)B * 8 * 96 * NQ * 2, stream);
    hipMemsetAsync(KbCA, 0, (size_t)B * 8 * 96 * 96 * 2, stream);
    hipMemsetAsync(VtCA, 0, (size_t)B * 8 * 96 * 96 * 2, stream);

    // fp32 -> bf16 activations
    cvt_f32_bf16_kernel<<<(T * D + 255) / 256, 256, 0, stream>>>(x, xb, T * D);
    cvt_f32_bf16_kernel<<<(TC * CD + 255) / 256, 256, 0, stream>>>(ctx, ctxb, TC * CD);

    // weight transpose+convert: W[K][N] -> Wt[N][K] bf16
    auto wt = [&](const float* W, u16* Wt, int K, int N) {
        wt_kernel<<<(K * N + 255) / 256, 256, 0, stream>>>(W, Wt, K, N);
    };
    wt(sa_wq, wsaq, D, D);
    wt(sa_wk, wsak, D, D);
    wt(sa_wv, wsav, D, D);
    wt(sa_wo, wsao, D, D);
    wt(ca_wq, wcaq, D, D);
    wt(ca_wk, wcak, CD, D);
    wt(ca_wv, wcav, CD, D);
    wt(ca_wo, wcao, D, D);
    wt(ff_w1, wff1, D, 2 * HID);
    wt(ff_w2, wff2, HID, D);

    auto gemm = [&](const u16* A, const u16* Bt, int M, int K, int N, int mode,
                    u16* ob, float* of, const float* bias, const float* resid,
                    int p0, int p1) {
        dim3 g((M + 63) / 64, N / 64);
        gemm_wmma_kernel<<<g, 128, 0, stream>>>(A, Bt, M, K, mode, ob, of, bias,
                                                resid, p0, p1);
    };

    // ---- self-attention ----
    gemm(xb, wsaq, T, D, D, 1, Qb, nullptr, nullptr, nullptr, NQ, NQ);
    gemm(xb, wsak, T, D, D, 1, KbSA, nullptr, nullptr, nullptr, NQ, NQ);
    gemm(xb, wsav, T, D, D, 2, VtSA, nullptr, nullptr, nullptr, NQ, NQ);
    attn_kernel<<<dim3(NQ / 16, B * 8), 32, 0, stream>>>(Qb, KbSA, VtSA, att, NQ, NQ,
                                                         scale);
    gemm(att, wsao, T, D, D, 3, nullptr, res, sa_bo, x, 0, 0);
    ln_kernel<<<T, 256, 0, stream>>>(res, ln1_g, ln1_b, ybuf, ybb);

    // ---- cross-attention ----
    gemm(ybb, wcaq, T, D, D, 1, Qb, nullptr, nullptr, nullptr, NQ, NQ);
    gemm(ctxb, wcak, TC, CD, D, 1, KbCA, nullptr, nullptr, nullptr, MC, 96);
    gemm(ctxb, wcav, TC, CD, D, 2, VtCA, nullptr, nullptr, nullptr, MC, 96);
    attn_kernel<<<dim3(NQ / 16, B * 8), 32, 0, stream>>>(Qb, KbCA, VtCA, att, 96, MC,
                                                         scale);
    gemm(att, wcao, T, D, D, 3, nullptr, res, ca_bo, ybuf, 0, 0);
    ln_kernel<<<T, 256, 0, stream>>>(res, ln2_g, ln2_b, ybuf, ybb);

    // ---- GEGLU feed-forward (gate first, then a-half multiplies in place) ----
    gemm(ybb, wff1 + (size_t)HID * D, T, D, HID, 4, gbuf, nullptr, ff_b1 + HID,
         nullptr, 0, 0);
    gemm(ybb, wff1, T, D, HID, 5, gbuf, nullptr, ff_b1, nullptr, 0, 0);
    gemm(gbuf, wff2, T, HID, D, 3, nullptr, res, ff_b2, ybuf, 0, 0);
    ln_kernel<<<T, 256, 0, stream>>>(res, ln3_g, ln3_b, (float*)d_out, nullptr);
}
